// Classifier_75316546502978
// MI455X (gfx1250) — matrix operand order repaired and checked by reading, not compiled, not run on hardware
//
#include <hip/hip_runtime.h>

#define B_TOT 256
#define S_LEN 900
#define I_DIM 50
#define E_DIM 200
#define H_DIM 200
#define G_DIM 800            // 4*H
#define KPAD  256            // 200 (h) + 50 (x_t) + 6 zero pad
#define NT    50             // N tiles (800/16)
#define KS    8              // K steps (256/32)
#define WAVES 10
#define THREADS 320          // 10 wave32
#define TILES_PER_WAVE 5     // 50 / 10
#define AP_STRIDE 264        // Apanel row stride in bf16 (132 words -> no bank conflicts)
#define GSTR 804             // gates row stride in floats (breaks half-wave bank aliasing)

typedef __attribute__((ext_vector_type(16))) __bf16 v16bf;
typedef __attribute__((ext_vector_type(8)))  __bf16 v8bf;
typedef __attribute__((ext_vector_type(8)))  float  v8f;

__device__ inline unsigned short f2bf(float x) {
  union { float f; unsigned u; } v; v.f = x;
  unsigned r = v.u + 0x7fffu + ((v.u >> 16) & 1u);   // round-to-nearest-even
  return (unsigned short)(r >> 16);
}
__device__ inline float sigmoidf_(float x) { return 1.f / (1.f + __expf(-x)); }
__device__ inline float tanhf_(float x) {           // fast tanh via exp
  float e = __expf(2.f * x);
  return 1.f - 2.f / (e + 1.f);
}

// ---------------------------------------------------------------------------
// Precompute: pack Bcat = [W_hh.T ; (W_ih@W_emb).T ; 0] (Kpad=256 x 800) into
// bf16 WMMA-fragment-linear layout, and fuse all biases:
//   bias_total[j] = b_hh[j] + b_ih[j] + sum_e W_ih[j,e]*b_emb[e]
// ---------------------------------------------------------------------------
__global__ void precompute_kernel(const float* __restrict__ W_emb,
                                  const float* __restrict__ b_emb,
                                  const float* __restrict__ W_ih,
                                  const float* __restrict__ b_ih,
                                  const float* __restrict__ W_hh,
                                  const float* __restrict__ b_hh,
                                  unsigned short* __restrict__ Bfrag,
                                  float* __restrict__ bias_total) {
  int gid = blockIdx.x * blockDim.x + threadIdx.x;
  const int NB = NT * KS * 32 * 16;                  // 204800 fragment elements
  if (gid < NB) {
    int e     = gid & 15;
    int lane  = (gid >> 4) & 31;
    int kstep = (gid >> 9) & 7;
    int ntile = gid >> 12;
    int j = ntile * 16 + (lane & 15);                // gate column (N)
    int k = kstep * 32 + (lane >> 4) * 8 + (e & 7) + (e >> 3) * 16; // reduction index (K)
    float v = 0.f;
    if (k < H_DIM) {
      v = W_hh[j * H_DIM + k];                       // recurrent weights
    } else if (k < H_DIM + I_DIM) {
      int kk = k - H_DIM;                            // Wc = W_ih @ W_emb
      float s = 0.f;
      for (int e2 = 0; e2 < E_DIM; ++e2)
        s += W_ih[j * E_DIM + e2] * W_emb[e2 * I_DIM + kk];
      v = s;
    }                                                // else zero pad
    Bfrag[gid] = f2bf(v);
  } else if (gid < NB + G_DIM) {
    int j = gid - NB;
    float s = b_hh[j] + b_ih[j];
    for (int e2 = 0; e2 < E_DIM; ++e2)
      s += W_ih[j * E_DIM + e2] * b_emb[e2];
    bias_total[j] = s;
  }
}

// ---------------------------------------------------------------------------
// Persistent LSTM: one workgroup per 16 batch rows; h/c live in LDS for all
// 900 steps; per step one fused WMMA GEMM gates = [h | x_t] @ Bcat.
// B streams from L2 each step; a laundered *offset* (not pointer) blocks LICM
// while preserving global address-space lowering; B is double-buffered per
// N-tile so each tile pays one L2 latency overlapped with 8 WMMAs.
// ---------------------------------------------------------------------------
__global__ __launch_bounds__(THREADS, 1)
void lstm_persistent(const float* __restrict__ src,
                     const unsigned short* __restrict__ Bfrag,
                     const float* __restrict__ bias_total,
                     const float* __restrict__ W_fc,
                     const float* __restrict__ b_fc,
                     float* __restrict__ out) {
  __shared__ __align__(16) unsigned short Apanel[16][AP_STRIDE]; // bf16 [h | x_t | 0]
  __shared__ float gatesL[16][GSTR];
  __shared__ float cpan[16][H_DIM];
  __shared__ float hf32[16][H_DIM];
  __shared__ float biasL[G_DIM];

  const int tid = threadIdx.x;
  const int rowbase = blockIdx.x * 16;

  // ---- init: h = c = 0, bias panel, x_0 ----
  for (int idx = tid; idx < 16 * AP_STRIDE; idx += THREADS) (&Apanel[0][0])[idx] = 0;
  for (int idx = tid; idx < 16 * H_DIM; idx += THREADS) {
    (&cpan[0][0])[idx] = 0.f; (&hf32[0][0])[idx] = 0.f;
  }
  for (int idx = tid; idx < G_DIM; idx += THREADS) biasL[idx] = bias_total[idx];
  for (int idx = tid; idx < 16 * I_DIM; idx += THREADS) {
    int b = idx / I_DIM, kk = idx - b * I_DIM;
    Apanel[b][H_DIM + kk] = f2bf(src[((size_t)(rowbase + b) * S_LEN) * I_DIM + kk]);
  }
  __syncthreads();

  const int wave = tid >> 5;
  const int lane = tid & 31;
  const int m    = lane & 15;
  const int half = lane >> 4;

  for (int t = 0; t < S_LEN; ++t) {
    // Launder an offset (value 0) so the 900-iteration loop cannot hoist the
    // loop-invariant B loads (would need 320 VGPRs -> scratch spills), while
    // keeping pointer provenance => global_load (not flat_load) lowering.
    size_t lz = 0;
    asm volatile("" : "+s"(lz));
    const unsigned short* bbase = Bfrag + lz;

    // ---- A fragments for this step (LDS -> 64 VGPRs) ----
    v16bf afrag[KS];
#pragma unroll
    for (int ks = 0; ks < KS; ++ks) {
      int k0 = ks * 32 + half * 8;
      v8bf lo = *(const v8bf*)&Apanel[m][k0];
      v8bf hi = *(const v8bf*)&Apanel[m][k0 + 16];
      afrag[ks] = __builtin_shufflevector(lo, hi,
                    0,1,2,3,4,5,6,7,8,9,10,11,12,13,14,15);
    }

    // ---- WMMA GEMM: gates[16,800] = A[16,256] @ Bcat[256,800] ----
    // Double-buffered B: load tile tt+1 while computing tile tt.
    v16bf bbuf[2][KS];
    {
      const unsigned short* bp =
          bbase + ((size_t)(wave * TILES_PER_WAVE) * KS * 32 + lane) * 16;
#pragma unroll
      for (int ks = 0; ks < KS; ++ks)
        bbuf[0][ks] = *(const v16bf*)(bp + (size_t)ks * 32 * 16);
    }
#pragma unroll
    for (int tt = 0; tt < TILES_PER_WAVE; ++tt) {
      const int cur = tt & 1;
      if (tt + 1 < TILES_PER_WAVE) {
        const unsigned short* bp =
            bbase + ((size_t)(wave * TILES_PER_WAVE + tt + 1) * KS * 32 + lane) * 16;
#pragma unroll
        for (int ks = 0; ks < KS; ++ks)
          bbuf[cur ^ 1][ks] = *(const v16bf*)(bp + (size_t)ks * 32 * 16);
      }
      v8f acc = {0.f,0.f,0.f,0.f,0.f,0.f,0.f,0.f};
#pragma unroll
      for (int ks = 0; ks < KS; ++ks)
        acc = __builtin_amdgcn_wmma_f32_16x16x32_bf16(
                  false, afrag[ks], false, bbuf[cur][ks], (short)0, acc, false, false);
      int j = (wave * TILES_PER_WAVE + tt) * 16 + m;
#pragma unroll
      for (int r = 0; r < 8; ++r) gatesL[r + half * 8][j] = acc[r];
    }
    __syncthreads();

    // ---- elementwise LSTM cell (f32) + prefetch x_{t+1} into Apanel ----
    for (int idx = tid; idx < 16 * H_DIM; idx += THREADS) {
      int b = idx / H_DIM, hh = idx - b * H_DIM;
      float gi = gatesL[b][hh]             + biasL[hh];
      float gf = gatesL[b][H_DIM + hh]     + biasL[H_DIM + hh];
      float gg = gatesL[b][2 * H_DIM + hh] + biasL[2 * H_DIM + hh];
      float go = gatesL[b][3 * H_DIM + hh] + biasL[3 * H_DIM + hh];
      float i = sigmoidf_(gi), f = sigmoidf_(gf), g = tanhf_(gg), o = sigmoidf_(go);
      float c = f * cpan[b][hh] + i * g;
      cpan[b][hh] = c;
      float h = o * tanhf_(c);
      hf32[b][hh] = h;
      Apanel[b][hh] = f2bf(h);                       // A operand for next step
    }
    if (t + 1 < S_LEN) {
      for (int idx = tid; idx < 16 * I_DIM; idx += THREADS) {
        int b = idx / I_DIM, kk = idx - b * I_DIM;
        Apanel[b][H_DIM + kk] =
            f2bf(src[((size_t)(rowbase + b) * S_LEN + (t + 1)) * I_DIM + kk]);
      }
    }
    __syncthreads();
  }

  // ---- outputs: features (f32 h_final) then tiny FC logits ----
  for (int idx = tid; idx < 16 * H_DIM; idx += THREADS) {
    int b = idx / H_DIM, hh = idx - b * H_DIM;
    out[B_TOT * 3 + (size_t)(rowbase + b) * H_DIM + hh] = hf32[b][hh];
  }
  if (tid < 48) {
    int b = tid / 3, cc = tid - (tid / 3) * 3;
    float s = b_fc[cc];
    for (int k = 0; k < H_DIM; ++k) s += hf32[b][k] * W_fc[cc * H_DIM + k];
    out[(size_t)(rowbase + b) * 3 + cc] = s;
  }
}

// ---------------------------------------------------------------------------
extern "C" void kernel_launch(void* const* d_in, const int* in_sizes, int n_in,
                              void* d_out, int out_size, void* d_ws, size_t ws_size,
                              hipStream_t stream) {
  const float* src   = (const float*)d_in[0];
  const float* W_emb = (const float*)d_in[1];
  const float* b_emb = (const float*)d_in[2];
  const float* W_ih  = (const float*)d_in[3];
  const float* b_ih  = (const float*)d_in[4];
  const float* W_hh  = (const float*)d_in[5];
  const float* b_hh  = (const float*)d_in[6];
  const float* W_fc  = (const float*)d_in[7];
  const float* b_fc  = (const float*)d_in[8];
  float* out = (float*)d_out;

  unsigned short* Bfrag = (unsigned short*)d_ws;                    // 409600 B
  float* bias_total = (float*)((char*)d_ws + (size_t)NT * KS * 32 * 16 * sizeof(unsigned short));

  const int npre = NT * KS * 32 * 16 + G_DIM;
  precompute_kernel<<<(npre + 255) / 256, 256, 0, stream>>>(
      W_emb, b_emb, W_ih, b_ih, W_hh, b_hh, Bfrag, bias_total);
  lstm_persistent<<<B_TOT / 16, THREADS, 0, stream>>>(
      src, Bfrag, bias_total, W_fc, b_fc, out);
}
